// AGaLiTeAttentionLayer_67010079752754
// MI455X (gfx1250) — matrix operand order repaired and checked by reading, not compiled, not run on hardware
//
#include <hip/hip_runtime.h>
#include <stdint.h>

typedef __attribute__((ext_vector_type(16))) _Float16 v16h;
typedef __attribute__((ext_vector_type(8)))  _Float16 v8h;
typedef __attribute__((ext_vector_type(8)))  float    v8f;

static constexpr int T_DIM  = 256;
static constexpr int B_DIM  = 32;
static constexpr int D_DIM  = 512;
static constexpr int H_DIM  = 8;
static constexpr int HD_DIM = 64;
static constexpr int ETA    = 4;
static constexpr int R_DIM  = 4;
static constexpr int FEAT   = HD_DIM * ETA;        // 256
static constexpr int M_ROWS = T_DIM * B_DIM;       // 8192
static constexpr int KDIM   = D_DIM;               // 512
static constexpr int NCAT   = 5 * 512 + 3 * 32;    // 2656 fused projection columns

// ---------------------------------------------------------------------------
// f32 -> f16 conversion (inputs / weights packing)
// ---------------------------------------------------------------------------
__global__ void cvt_f32_f16(const float* __restrict__ src, _Float16* __restrict__ dst, int n) {
  int i = blockIdx.x * blockDim.x + threadIdx.x;
  if (i < n) dst[i] = (_Float16)src[i];
}

// ---------------------------------------------------------------------------
// WMMA GEMM, 64x32 macro-tile per wave (4 M-subtiles x 2 N-subtiles).
// C[M x ldc] = A[M x K] * Bw[N x K]^T (+ bias); f16 in, f32 accumulate/out.
// 12 b128 loads per 8 v_wmma (vs 4 per 1 unblocked). Grid sized so
// wave-groups == macro-tiles exactly -> EXEC all-ones at every WMMA.
// ---------------------------------------------------------------------------
__global__ void __launch_bounds__(256) wmma_gemm(
    const _Float16* __restrict__ A,    // M x K row-major
    const _Float16* __restrict__ Bw,   // N x K row-major (weight, used as B^T)
    float* __restrict__ C,             // M x ldc row-major
    const float* __restrict__ bias,    // length >= N, or nullptr
    int NB2, int K, int ldc)           // NB2 = (N/16)/2 macro-columns
{
  const int wave = threadIdx.x >> 5;
  const int lane = threadIdx.x & 31;
  const long g   = (long)blockIdx.x * 8 + wave;
  const int  gM  = (int)(g / NB2);
  const int  gN  = (int)(g % NB2);
  const int  m0  = gM * 64, n0 = gN * 32;
  const int  hi  = lane >> 4;     // half-wave select
  const int  lm  = lane & 15;

  const _Float16* arow[4];
  const _Float16* brow[2];
#pragma unroll
  for (int mi = 0; mi < 4; ++mi) arow[mi] = A  + (long)(m0 + mi * 16 + lm) * K + hi * 8;
#pragma unroll
  for (int nj = 0; nj < 2; ++nj) brow[nj] = Bw + (long)(n0 + nj * 16 + lm) * K + hi * 8;

  v8f acc[4][2] = {};
  for (int k = 0; k < K; k += 32) {
#pragma unroll
    for (int mi = 0; mi < 4; ++mi) __builtin_prefetch(arow[mi] + k + 64, 0, 1);
#pragma unroll
    for (int nj = 0; nj < 2; ++nj) __builtin_prefetch(brow[nj] + k + 64, 0, 1);
    v16h afr[4], bfr[2];
#pragma unroll
    for (int mi = 0; mi < 4; ++mi) {
      v8h lo = *(const v8h*)(arow[mi] + k);        // K = k+hi*8    .. +7
      v8h hh = *(const v8h*)(arow[mi] + k + 16);   // K = k+16+hi*8 .. +7
#pragma unroll
      for (int j = 0; j < 8; ++j) { afr[mi][j] = lo[j]; afr[mi][j + 8] = hh[j]; }
    }
#pragma unroll
    for (int nj = 0; nj < 2; ++nj) {
      v8h lo = *(const v8h*)(brow[nj] + k);        // K = k+hi*8    .. +7
      v8h hh = *(const v8h*)(brow[nj] + k + 16);   // K = k+16+hi*8 .. +7
#pragma unroll
      for (int j = 0; j < 8; ++j) { bfr[nj][j] = lo[j]; bfr[nj][j + 8] = hh[j]; }
    }
#pragma unroll
    for (int mi = 0; mi < 4; ++mi)
#pragma unroll
      for (int nj = 0; nj < 2; ++nj)
        acc[mi][nj] = __builtin_amdgcn_wmma_f32_16x16x32_f16(
            false, afr[mi], false, bfr[nj], (short)0, acc[mi][nj], false, false);
  }

#pragma unroll
  for (int mi = 0; mi < 4; ++mi)
#pragma unroll
    for (int nj = 0; nj < 2; ++nj) {
      const int nc = n0 + nj * 16 + lm;
      const float bv = bias ? bias[nc] : 0.0f;
      float* cp = C + (long)(m0 + mi * 16 + hi * 8) * ldc + nc;
#pragma unroll
      for (int j = 0; j < 8; ++j) cp[(long)j * ldc] = acc[mi][nj][j] + bv;
    }
}

// ---------------------------------------------------------------------------
// ds_swizzle butterfly all-reduce within a wave32 (no barriers).
// Group-of-32 XOR swizzles: SWAPX16/8/4/2/1.
// ---------------------------------------------------------------------------
__device__ __forceinline__ float wave_allred(float x) {
  x += __builtin_bit_cast(float, __builtin_amdgcn_ds_swizzle(__builtin_bit_cast(int, x), 0x401F));
  x += __builtin_bit_cast(float, __builtin_amdgcn_ds_swizzle(__builtin_bit_cast(int, x), 0x201F));
  x += __builtin_bit_cast(float, __builtin_amdgcn_ds_swizzle(__builtin_bit_cast(int, x), 0x101F));
  x += __builtin_bit_cast(float, __builtin_amdgcn_ds_swizzle(__builtin_bit_cast(int, x), 0x081F));
  x += __builtin_bit_cast(float, __builtin_amdgcn_ds_swizzle(__builtin_bit_cast(int, x), 0x041F));
  return x;
}

__device__ __forceinline__ float sigmoidf_(float x) { return 1.0f / (1.0f + __expf(-x)); }

// ---------------------------------------------------------------------------
// Fused feature-map + oscillator + 3 discounted scans + attention contraction.
// One block per (b,h); 256 threads; thread = feature F = e*64+d (e == r).
// Recurrent states live in REGISTERS (thread-private across t).
// Per-t Y-row slice is double-buffered into LDS with global_load_async_to_lds
// (ASYNCcnt), issued one step ahead; 3 barriers per time step.
// ---------------------------------------------------------------------------
__global__ void __launch_bounds__(256) agalite_scan(
    const float* __restrict__ Y,        // M_ROWS x NCAT (fused projections)
    const int*   __restrict__ term,     // T x B
    const float* __restrict__ k_prev,   // B,R,H,FEAT
    const float* __restrict__ v_prev,   // B,R,H,HD
    const float* __restrict__ s_prev,   // B,H,FEAT
    const float* __restrict__ tick,     // B
    _Float16* __restrict__ attn16)      // M_ROWS x (H*HD)
{
  const int b   = blockIdx.x >> 3;
  const int h   = blockIdx.x & 7;
  const int tid = threadIdx.x;
  const int e   = tid >> 6;   // eta index (== r index for the value partition)
  const int d   = tid & 63;
  const int wid = tid >> 5;

  // stage layout (floats): [0..63]=q [64..127]=k [128..191]=v
  // [192..255]=beta_raw [256..319]=gamma_raw [320..323]=p1 [324..327]=p2 [328..331]=p3_raw
  __shared__ float stage[2][336];
  __shared__ float wred[5][8];
  __shared__ float redkv[256];
  __shared__ float tmask[T_DIM];

  auto issue_stage = [&](int tt, int buf) {
    if (tid < 83) {
      const long rowoff = ((long)tt * B_DIM + b) * NCAT;
      const float* src;
      int slot;
      if (tid < 80) {             // q,k,v,beta,gamma: 5 arrays x 64 floats, 16B chunks
        const int a = tid >> 4, l = tid & 15;
        src  = Y + rowoff + a * 512 + h * 64 + l * 4;
        slot = a * 64 + l * 4;
      } else {                    // p1,p2,p3: 4 floats each, stride 32 cols
        const int a = tid - 80;
        src  = Y + rowoff + 2560 + a * 32 + h * ETA;
        slot = 320 + a * 4;
      }
      unsigned laddr = (unsigned)(uintptr_t)&stage[buf][slot];
      asm volatile("global_load_async_to_lds_b128 %0, %1, off"
                   :: "v"(laddr), "v"(src) : "memory");
    }
  };

  // prefetch t=0 stage while we load initial state
  issue_stage(0, 0);

  // terminations for this b, all t (LDS broadcast later)
  tmask[tid] = 1.0f - (float)term[tid * B_DIM + b];

  // ---- recurrent state in registers ----
  float sk[R_DIM];
#pragma unroll
  for (int r = 0; r < R_DIM; ++r)
    sk[r] = k_prev[(((long)b * R_DIM + r) * H_DIM + h) * FEAT + tid];
  float ssr = s_prev[((long)b * H_DIM + h) * FEAT + tid];
  float svr = v_prev[(((long)b * R_DIM + e) * H_DIM + h) * HD_DIM + d];

  // ---- oscillators: cos((tick + t + 1) * omega_r), omega_r = -pi + r*2pi/3 ----
  const float PI = 3.14159265358979323846f;
  const float tk = tick[b];
  float cr[R_DIM], srv[R_DIM], cw[R_DIM], sw[R_DIM];
#pragma unroll
  for (int r = 0; r < R_DIM; ++r) {
    float w = -PI + (float)r * (2.0f * PI / 3.0f);
    cw[r] = __cosf(w); sw[r] = __sinf(w);
    float a0 = (tk + 1.0f) * w;
    cr[r] = __cosf(a0); srv[r] = __sinf(a0);
  }

  asm volatile("s_wait_asynccnt 0x0" ::: "memory");
  __syncthreads();

  for (int t = 0; t < T_DIM; ++t) {
    const int buf = t & 1;
    if (t + 1 < T_DIM) issue_stage(t + 1, buf ^ 1);   // overlap next row fetch

    const float* st = stage[buf];
    const float qd  = st[d];
    const float kk  = st[64 + d];
    const float vvd = st[128 + d];
    const float be  = sigmoidf_(st[192 + d]);
    const float ga  = sigmoidf_(st[256 + d]);
    const float p1e = st[320 + e];
    const float p2e = st[324 + e];
    const float p3e = sigmoidf_(st[328 + e]);
    const float mask = tmask[t];

    const float phiq = fmaxf(qd * p2e, 0.0f);
    const float psik = fmaxf(kk * p1e, 0.0f);
    const float gamf = ga * p3e;
    const float gk   = psik * gamf;
    const float dg   = (1.0f - gamf) * mask;

    ssr = dg * ssr + gk;                       // final_s scan
    float pr[5];
#pragma unroll
    for (int r = 0; r < R_DIM; ++r) {          // final_keys scan, all r per feature
      sk[r] = dg * sk[r] + gk * cr[r];
      pr[r] = sk[r] * phiq;
    }
    pr[4] = ssr * phiq;

    const float gv = vvd * be;                 // final_values scan (r == e)
    const float db = (1.0f - be) * mask;
    svr = db * svr + gv * cr[e];

    // 5 cross-FEAT reductions: wave-level butterflies, then 8-wave LDS combine
#pragma unroll
    for (int a = 0; a < 5; ++a) pr[a] = wave_allred(pr[a]);
    if ((tid & 31) == 0) {
#pragma unroll
      for (int a = 0; a < 5; ++a) wred[a][wid] = pr[a];
    }
    __syncthreads();                           // (A)
    float kdq_r = 0.0f, norm = 0.0f;
#pragma unroll
    for (int w8 = 0; w8 < 8; ++w8) { kdq_r += wred[e][w8]; norm += wred[4][w8]; }

    redkv[tid] = svr * kdq_r;                  // kv partial: sum over r
    __syncthreads();                           // (B)
    if (tid < 64) {
      const float kvd = redkv[tid] + redkv[64 + tid] + redkv[128 + tid] + redkv[192 + tid];
      const float attn = kvd / (8.0f * norm + 1e-6f);   // 2*R = 8
      const long row = (long)t * B_DIM + b;
      attn16[row * (H_DIM * HD_DIM) + h * HD_DIM + tid] = (_Float16)attn;
    }

#pragma unroll
    for (int r = 0; r < R_DIM; ++r) {          // advance oscillators by omega_r
      const float c2 = cr[r] * cw[r] - srv[r] * sw[r];
      const float s2 = srv[r] * cw[r] + cr[r] * sw[r];
      cr[r] = c2; srv[r] = s2;
    }

    asm volatile("s_wait_asynccnt 0x0" ::: "memory");   // t+1 stage landed
    __syncthreads();                           // (C) also fences redkv/wred reuse
  }
}

// ---------------------------------------------------------------------------
extern "C" void kernel_launch(void* const* d_in, const int* in_sizes, int n_in,
                              void* d_out, int out_size, void* d_ws, size_t ws_size,
                              hipStream_t stream) {
  const float* inputs = (const float*)d_in[0];
  const int*   term   = (const int*)  d_in[1];
  const float* kprev  = (const float*)d_in[2];
  const float* vprev  = (const float*)d_in[3];
  const float* sprev  = (const float*)d_in[4];
  const float* tick   = (const float*)d_in[5];
  const float* Wq  = (const float*)d_in[6];
  const float* Wk  = (const float*)d_in[7];
  const float* Wv  = (const float*)d_in[8];
  const float* Wb  = (const float*)d_in[9];
  const float* Wg  = (const float*)d_in[10];
  const float* Wp1 = (const float*)d_in[11];
  const float* Wp2 = (const float*)d_in[12];
  const float* Wp3 = (const float*)d_in[13];
  const float* Wo  = (const float*)d_in[14];
  const float* bo  = (const float*)d_in[15];
  (void)in_sizes; (void)n_in; (void)out_size; (void)ws_size;

  char* ws = (char*)d_ws;
  size_t off = 0;
  auto carve = [&](size_t bytes) -> void* {
    void* p = ws + off;
    off = (off + bytes + 255) & ~(size_t)255;
    return p;
  };
  _Float16* X16    = (_Float16*)carve((size_t)M_ROWS * KDIM * 2);   // 8 MB
  _Float16* Wcat   = (_Float16*)carve((size_t)NCAT  * KDIM * 2);    // 2.7 MB
  _Float16* Wo16   = (_Float16*)carve((size_t)D_DIM * KDIM * 2);    // 0.5 MB
  float*    Yb     = (float*)   carve((size_t)M_ROWS * NCAT * 4);   // 87 MB (L2-resident)
  _Float16* attn16 = (_Float16*)carve((size_t)M_ROWS * D_DIM * 2);  // 8 MB

  auto cvt = [&](const float* s, _Float16* dgt, int n) {
    cvt_f32_f16<<<(n + 255) / 256, 256, 0, stream>>>(s, dgt, n);
  };
  cvt(inputs, X16, M_ROWS * KDIM);
  cvt(Wq,  Wcat + (size_t)0 * 512 * 512, 512 * 512);
  cvt(Wk,  Wcat + (size_t)1 * 512 * 512, 512 * 512);
  cvt(Wv,  Wcat + (size_t)2 * 512 * 512, 512 * 512);
  cvt(Wb,  Wcat + (size_t)3 * 512 * 512, 512 * 512);
  cvt(Wg,  Wcat + (size_t)4 * 512 * 512, 512 * 512);
  cvt(Wp1, Wcat + (size_t)5 * 512 * 512,             32 * 512);
  cvt(Wp2, Wcat + (size_t)5 * 512 * 512 + 32 * 512,  32 * 512);
  cvt(Wp3, Wcat + (size_t)5 * 512 * 512 + 64 * 512,  32 * 512);
  cvt(Wo,  Wo16, 512 * 512);

  {
    const int NB2 = (NCAT / 16) / 2;                    // 83 macro-columns
    const int groups = (M_ROWS / 64) * NB2;             // 128 * 83 = 10624
    wmma_gemm<<<groups / 8, 256, 0, stream>>>(X16, Wcat, Yb, nullptr, NB2, KDIM, NCAT);
  }

  agalite_scan<<<B_DIM * H_DIM, 256, 0, stream>>>(Yb, term, kprev, vprev, sprev, tick, attn16);

  {
    const int NB2 = (D_DIM / 16) / 2;                   // 16 macro-columns
    const int groups = (M_ROWS / 64) * NB2;             // 128 * 16 = 2048
    wmma_gemm<<<groups / 8, 256, 0, stream>>>(attn16, Wo16, (float*)d_out, bo, NB2, KDIM, D_DIM);
  }
}
